// RelPosAttention_90417651515525
// MI455X (gfx1250) — matrix operand order repaired and checked by reading, not compiled, not run on hardware
//
#include <hip/hip_runtime.h>
#include <stdint.h>

typedef unsigned short u16;
typedef __bf16 bf16x16 __attribute__((ext_vector_type(16)));
typedef float  f32x8   __attribute__((ext_vector_type(8)));

union ABFrag { uint4 q[2]; bf16x16 v; };
union V16u   { uint4 q[2]; u16 s[16]; };

__device__ inline u16 f2bf(float f) {
    union { float f; uint32_t u; } c; c.f = f;
    uint32_t u = c.u;
    u += 0x7fffu + ((u >> 16) & 1u);   // round-to-nearest-even
    return (u16)(u >> 16);
}

__device__ inline f32x8 wmma_bf16(const ABFrag& a, const ABFrag& b, f32x8 c) {
    return __builtin_amdgcn_wmma_f32_16x16x32_bf16(
        /*neg_a=*/false, a.v, /*neg_b=*/false, b.v,
        /*c_mod=*/(short)0, c, /*reuse_a=*/false, /*reuse_b=*/false);
}

#define D_MODEL 1024
#define N_HEAD  16
#define D_HEAD  64
#define MAX_DIST 128
#define BSZ 4
#define LSEQ 2048
#define ROWS (BSZ * LSEQ)      // 8192
#define NQKV (3 * D_MODEL)     // 3072

// ---------------- conversion kernels ----------------
__global__ void k_f32_to_bf16(const float* __restrict__ in, u16* __restrict__ out, int n) {
    int i = blockIdx.x * blockDim.x + threadIdx.x;
    if (i < n) out[i] = f2bf(in[i]);
}

// in[K][N] (row-major) -> out[N][K] bf16
__global__ void k_transpose_bf16(const float* __restrict__ in, u16* __restrict__ out,
                                 int K, int N) {
    int i = blockIdx.x * blockDim.x + threadIdx.x;
    if (i >= K * N) return;
    int n = i / K, k = i - n * K;
    out[i] = f2bf(in[k * N + n]);
}

// ---------------- QKV projection GEMM ----------------
// X bf16 [8192 x 1024], Wt bf16 [3072 x 1024] (pre-transposed), bias f32 [3072]
// one wave -> 32x64 tile of C (2 A row-groups x 4 B tiles, B reused twice);
// scatter epilogue into Q/K/V [B,H,L,64] bf16
__global__ __launch_bounds__(256)
void k_gemm_qkv(const u16* __restrict__ X, const u16* __restrict__ Wt,
                const float* __restrict__ bias,
                u16* __restrict__ Qo, u16* __restrict__ Ko, u16* __restrict__ Vo) {
    const int NT64 = NQKV / 64;                   // 48
    int wv   = blockIdx.x * 8 + (threadIdx.x >> 5);
    int lane = threadIdx.x & 31;
    int hi = lane >> 4, lo = lane & 15;
    int mt = wv / NT64, nt = wv % NT64;           // mt in [0,256), 32 rows each

    const u16* arow0 = X + (size_t)(mt * 32 + lo) * D_MODEL;
    const u16* arow1 = arow0 + (size_t)16 * D_MODEL;
    f32x8 zero = {0.f,0.f,0.f,0.f,0.f,0.f,0.f,0.f};
    f32x8 acc[2][4] = {{zero, zero, zero, zero}, {zero, zero, zero, zero}};

    for (int kb = 0; kb < D_MODEL; kb += 32) {
        ABFrag a0, a1;
        a0.q[0] = *(const uint4*)(arow0 + kb + hi * 8);
        a0.q[1] = *(const uint4*)(arow0 + kb + hi * 8 + 16);
        a1.q[0] = *(const uint4*)(arow1 + kb + hi * 8);
        a1.q[1] = *(const uint4*)(arow1 + kb + hi * 8 + 16);
        #pragma unroll
        for (int j = 0; j < 4; ++j) {
            int col = nt * 64 + j * 16 + lo;
            const u16* brow = Wt + (size_t)col * D_MODEL + kb + hi * 8;
            ABFrag b;
            b.q[0] = *(const uint4*)(brow);
            b.q[1] = *(const uint4*)(brow + 16);
            acc[0][j] = wmma_bf16(a0, b, acc[0][j]);
            acc[1][j] = wmma_bf16(a1, b, acc[1][j]);
        }
    }
    #pragma unroll
    for (int j = 0; j < 4; ++j) {
        int col = nt * 64 + j * 16 + lo;
        float bs = bias[col];
        int seg = col >> 10, rem = col & 1023;
        int h = rem >> 6, d = rem & 63;
        u16* dst = (seg == 0) ? Qo : (seg == 1) ? Ko : Vo;
        #pragma unroll
        for (int g = 0; g < 2; ++g) {
            #pragma unroll
            for (int r = 0; r < 8; ++r) {
                int row = mt * 32 + g * 16 + hi * 8 + r;
                int b = row >> 11, tok = row & 2047;
                dst[((size_t)((b * N_HEAD + h) * LSEQ + tok)) * D_HEAD + d] =
                    f2bf(acc[g][j][r] + bs);
            }
        }
    }
}

// ---------------- flash attention ----------------
// block = 128 threads (4 waves), each wave owns 16 query rows of one (b,h).
__global__ __launch_bounds__(128)
void k_attn(const u16* __restrict__ Q, const u16* __restrict__ K,
            const u16* __restrict__ V, const float* __restrict__ relbias,
            const unsigned char* __restrict__ kpm, u16* __restrict__ Ctx) {
    __shared__ __align__(16) float relb[2 * MAX_DIST + 1];
    __shared__ __align__(16) u16 Pt[4][16 * 40];
    __shared__ __align__(16) u16 Vt[64 * 40];

    const int tid = threadIdx.x;
    const int w = tid >> 5, lane = tid & 31;
    const int hi = lane >> 4, lo = lane & 15;
    const int bh = blockIdx.y;
    const int b = bh >> 4, h = bh & 15;
    const int q0 = blockIdx.x * 64 + w * 16;
    const float LOG2E = 1.44269504088896340f;
    const float SCALE = 0.125f;   // 1/sqrt(64)

    for (int i = tid; i < 2 * MAX_DIST + 1; i += 128)
        relb[i] = relbias[h * (2 * MAX_DIST + 1) + i];

    const size_t bhbase = (size_t)bh * LSEQ * D_HEAD;
    const u16* qptr = Q + bhbase + (size_t)(q0 + lo) * D_HEAD;
    ABFrag qa0, qa1;                       // A-fragments for d 0..31 / 32..63
    qa0.q[0] = *(const uint4*)(qptr + hi * 8);
    qa0.q[1] = *(const uint4*)(qptr + hi * 8 + 16);
    qa1.q[0] = *(const uint4*)(qptr + 32 + hi * 8);
    qa1.q[1] = *(const uint4*)(qptr + 32 + hi * 8 + 16);

    f32x8 zero = {0.f,0.f,0.f,0.f,0.f,0.f,0.f,0.f};
    f32x8 acc[4] = {zero, zero, zero, zero};
    float m_run[8], l_run[8];
    #pragma unroll
    for (int r = 0; r < 8; ++r) { m_run[r] = -__builtin_inff(); l_run[r] = 0.f; }

    __syncthreads();   // relb ready

    for (int key0 = 0; key0 < LSEQ; key0 += 32) {
        // cooperative V-chunk transpose into LDS: Vt[d][kc], stride 40
        {
            int kc = tid & 31, dseg = tid >> 5;
            const u16* vp = V + bhbase + (size_t)(key0 + kc) * D_HEAD + dseg * 16;
            V16u t;
            t.q[0] = *(const uint4*)(vp);
            t.q[1] = *(const uint4*)(vp + 8);
            #pragma unroll
            for (int j = 0; j < 16; ++j) Vt[(dseg * 16 + j) * 40 + kc] = t.s[j];
        }
        if (key0 + 32 < LSEQ) {            // global_prefetch_b8 for next chunk
            __builtin_prefetch(K + bhbase + (size_t)(key0 + 32 + lane) * D_HEAD, 0, 0);
            __builtin_prefetch(V + bhbase + (size_t)(key0 + 32 + lane) * D_HEAD, 0, 0);
        }

        // S = (Q K^T): K rows are exactly the B-fragment layout (k=d contiguous)
        f32x8 S[2];
        #pragma unroll
        for (int nt = 0; nt < 2; ++nt) {
            const u16* kp = K + bhbase + (size_t)(key0 + nt * 16 + lo) * D_HEAD;
            ABFrag k0, k1;
            k0.q[0] = *(const uint4*)(kp + hi * 8);
            k0.q[1] = *(const uint4*)(kp + hi * 8 + 16);
            k1.q[0] = *(const uint4*)(kp + 32 + hi * 8);
            k1.q[1] = *(const uint4*)(kp + 32 + hi * 8 + 16);
            S[nt] = wmma_bf16(qa0, k0, zero);
            S[nt] = wmma_bf16(qa1, k1, S[nt]);
        }

        float masked[2];
        #pragma unroll
        for (int nt = 0; nt < 2; ++nt)
            masked[nt] = kpm[b * LSEQ + key0 + nt * 16 + lo] ? 0.f : 1.f;

        // online softmax (rows = r + 8*hi; 16-lane groups share a row)
        #pragma unroll
        for (int r = 0; r < 8; ++r) {
            int qr = q0 + hi * 8 + r;
            float sv[2];
            #pragma unroll
            for (int nt = 0; nt < 2; ++nt) {
                int key = key0 + nt * 16 + lo;
                int rp = key - qr;
                rp = rp < -MAX_DIST ? -MAX_DIST : (rp > MAX_DIST ? MAX_DIST : rp);
                float s = S[nt][r] * SCALE + relb[rp + MAX_DIST];
                sv[nt] = (masked[nt] != 0.f) ? -__builtin_inff() : s;
            }
            float cm = fmaxf(sv[0], sv[1]);
            #pragma unroll
            for (int d = 1; d < 16; d <<= 1) cm = fmaxf(cm, __shfl_xor(cm, d, 32));
            float nm = fmaxf(m_run[r], cm);
            float sf = (m_run[r] == -__builtin_inff())
                           ? 0.f : exp2f((m_run[r] - nm) * LOG2E);
            float psum = 0.f;
            #pragma unroll
            for (int nt = 0; nt < 2; ++nt) {
                float p = (sv[nt] == -__builtin_inff())
                              ? 0.f : exp2f((sv[nt] - nm) * LOG2E);
                psum += p;
                Pt[w][(hi * 8 + r) * 40 + nt * 16 + lo] = f2bf(p);
            }
            #pragma unroll
            for (int d = 1; d < 16; d <<= 1) psum += __shfl_xor(psum, d, 32);
            l_run[r] = l_run[r] * sf + psum;
            m_run[r] = nm;
            #pragma unroll
            for (int dt = 0; dt < 4; ++dt) acc[dt][r] *= sf;
        }
        __syncthreads();   // Pt + Vt visible

        // O += P @ V
        ABFrag pa;
        const u16* prow = &Pt[w][lo * 40 + hi * 8];
        pa.q[0] = *(const uint4*)(prow);
        pa.q[1] = *(const uint4*)(prow + 16);
        #pragma unroll
        for (int dt = 0; dt < 4; ++dt) {
            ABFrag vb;
            const u16* vrow = &Vt[(dt * 16 + lo) * 40 + hi * 8];
            vb.q[0] = *(const uint4*)(vrow);
            vb.q[1] = *(const uint4*)(vrow + 16);
            acc[dt] = wmma_bf16(pa, vb, acc[dt]);
        }
        __syncthreads();   // before next chunk overwrites Pt/Vt
    }

    // normalize + write context in [B,L,D_MODEL] (head-concatenated) bf16
    #pragma unroll
    for (int r = 0; r < 8; ++r) {
        float inv = l_run[r] > 0.f ? 1.f / l_run[r] : 0.f;
        int qr = q0 + hi * 8 + r;
        size_t rowbase = ((size_t)(b * LSEQ + qr)) * D_MODEL + h * D_HEAD;
        #pragma unroll
        for (int dt = 0; dt < 4; ++dt)
            Ctx[rowbase + dt * 16 + lo] = f2bf(acc[dt][r] * inv);
    }
}

// ---------------- output projection GEMM ----------------
// Ctx bf16 [8192 x 1024] @ Wo^T bf16 [1024 x 1024] + bo -> out f32
// one wave -> 32x64 tile, B fragments reused twice
__global__ __launch_bounds__(256)
void k_gemm_out(const u16* __restrict__ A, const u16* __restrict__ Wt,
                const float* __restrict__ bias, float* __restrict__ out) {
    const int NT64 = D_MODEL / 64;                // 16
    int wv   = blockIdx.x * 8 + (threadIdx.x >> 5);
    int lane = threadIdx.x & 31;
    int hi = lane >> 4, lo = lane & 15;
    int mt = wv / NT64, nt = wv % NT64;           // mt in [0,256)

    const u16* arow0 = A + (size_t)(mt * 32 + lo) * D_MODEL;
    const u16* arow1 = arow0 + (size_t)16 * D_MODEL;
    f32x8 zero = {0.f,0.f,0.f,0.f,0.f,0.f,0.f,0.f};
    f32x8 acc[2][4] = {{zero, zero, zero, zero}, {zero, zero, zero, zero}};

    for (int kb = 0; kb < D_MODEL; kb += 32) {
        ABFrag a0, a1;
        a0.q[0] = *(const uint4*)(arow0 + kb + hi * 8);
        a0.q[1] = *(const uint4*)(arow0 + kb + hi * 8 + 16);
        a1.q[0] = *(const uint4*)(arow1 + kb + hi * 8);
        a1.q[1] = *(const uint4*)(arow1 + kb + hi * 8 + 16);
        #pragma unroll
        for (int j = 0; j < 4; ++j) {
            int col = nt * 64 + j * 16 + lo;
            const u16* brow = Wt + (size_t)col * D_MODEL + kb + hi * 8;
            ABFrag b;
            b.q[0] = *(const uint4*)(brow);
            b.q[1] = *(const uint4*)(brow + 16);
            acc[0][j] = wmma_bf16(a0, b, acc[0][j]);
            acc[1][j] = wmma_bf16(a1, b, acc[1][j]);
        }
    }
    #pragma unroll
    for (int j = 0; j < 4; ++j) {
        int col = nt * 64 + j * 16 + lo;
        float bs = bias[col];
        #pragma unroll
        for (int g = 0; g < 2; ++g) {
            #pragma unroll
            for (int r = 0; r < 8; ++r) {
                int row = mt * 32 + g * 16 + hi * 8 + r;
                out[(size_t)row * D_MODEL + col] = acc[g][j][r] + bs;
            }
        }
    }
}

// ---------------- launcher ----------------
extern "C" void kernel_launch(void* const* d_in, const int* in_sizes, int n_in,
                              void* d_out, int out_size, void* d_ws, size_t ws_size,
                              hipStream_t stream) {
    const float* x            = (const float*)d_in[0];
    const unsigned char* kpm  = (const unsigned char*)d_in[1];  // bool mask
    const float* Wqkv         = (const float*)d_in[2];
    const float* bqkv         = (const float*)d_in[3];
    const float* Wo           = (const float*)d_in[4];
    const float* bo           = (const float*)d_in[5];
    const float* relb         = (const float*)d_in[6];
    float* out = (float*)d_out;

    char* ws = (char*)d_ws;
    u16* Xbf   = (u16*)(ws);                               // 16 MiB
    u16* Wqkvt = (u16*)(ws + (size_t)16777216);            //  6 MiB
    u16* Wot   = (u16*)(ws + (size_t)23068672);            //  2 MiB
    u16* Qb    = (u16*)(ws + (size_t)25165824);            // 16 MiB
    u16* Kb    = (u16*)(ws + (size_t)41943040);            // 16 MiB
    u16* Vb    = (u16*)(ws + (size_t)58720256);            // 16 MiB
    u16* Ctx   = (u16*)(ws + (size_t)75497472);            // 16 MiB

    { int n = ROWS * D_MODEL;
      k_f32_to_bf16<<<(n + 255) / 256, 256, 0, stream>>>(x, Xbf, n); }
    { int n = D_MODEL * NQKV;
      k_transpose_bf16<<<(n + 255) / 256, 256, 0, stream>>>(Wqkv, Wqkvt, D_MODEL, NQKV); }
    { int n = D_MODEL * D_MODEL;
      k_transpose_bf16<<<(n + 255) / 256, 256, 0, stream>>>(Wo, Wot, D_MODEL, D_MODEL); }

    // (8192/32) * (3072/64) waves / 8 waves per block = 1536 blocks
    k_gemm_qkv<<<1536, 256, 0, stream>>>(Xbf, Wqkvt, bqkv, Qb, Kb, Vb);

    // grid.x = L/64 query blocks, grid.y = B*H
    k_attn<<<dim3(LSEQ / 64, BSZ * N_HEAD), 128, 0, stream>>>(Qb, Kb, Vb, relb, kpm, Ctx);

    // (8192/32) * (1024/64) / 8 = 512 blocks
    k_gemm_out<<<512, 256, 0, stream>>>(Ctx, Wot, bo, out);
}
